// SubGraph_14362370638062
// MI455X (gfx1250) — matrix operand order repaired and checked by reading, not compiled, not run on hardware
//
#include <hip/hip_runtime.h>
#include <hip/hip_bf16.h>

typedef __attribute__((ext_vector_type(16))) _Float16 v16h;
typedef __attribute__((ext_vector_type(8)))  _Float16 v8h;
typedef __attribute__((ext_vector_type(8)))  float    v8f;

#define HC 128
#define NHEAD 4
#define CPH 32
#define NEG_SLOPE 0.2f

// ---------------------------------------------------------------------------
// one-shot converts
// ---------------------------------------------------------------------------
__global__ void f32_to_f16_kernel(const float* __restrict__ src,
                                  _Float16* __restrict__ dst, int n)
{
    int t = blockIdx.x * blockDim.x + threadIdx.x;
    if (t < n) dst[t] = (_Float16)src[t];
}

// W [K,HC] f32 row-major  ->  WT [HC,K] f16  (so B fragments load contiguously)
__global__ void w_transpose_f16_kernel(const float* __restrict__ W,
                                       _Float16* __restrict__ WT, int K)
{
    int t = blockIdx.x * blockDim.x + threadIdx.x;
    if (t >= K * HC) return;
    int k = t / HC, c = t % HC;
    WT[(size_t)c * K + k] = (_Float16)W[t];
}

// ---------------------------------------------------------------------------
// Dual GEMM: XL = X@Wl + bl, XR = X@Wr + br
// Xh: [n,K] f16 row-major; WlT/WrT: [HC,K] f16 (transposed weights)
// One wave per 16x16 output tile. All fragment loads are 16B v8h (b128).
// ---------------------------------------------------------------------------
__global__ __launch_bounds__(32)
void gatv2_dual_gemm(const _Float16* __restrict__ Xh,
                     const _Float16* __restrict__ WlT,
                     const _Float16* __restrict__ WrT,
                     const float* __restrict__ bl, const float* __restrict__ br,
                     float* __restrict__ XL, float* __restrict__ XR,
                     int K)
{
    const int row0 = blockIdx.x * 16;
    const int col0 = blockIdx.y * 16;
    const int lane = threadIdx.x;          // 0..31
    const int half = lane >> 4;            // 0 or 1
    const int lrow = lane & 15;            // 0..15

    const _Float16* xrow  = Xh  + (size_t)(row0 + lrow) * K;
    const _Float16* wlrow = WlT + (size_t)(col0 + lrow) * K;
    const _Float16* wrrow = WrT + (size_t)(col0 + lrow) * K;

    v8f accL = {};
    v8f accR = {};

    for (int k0 = 0; k0 < K; k0 += 32) {
        // A 16x32 f16 ISA layout: lane<16 K={k0..k0+7, k0+16..23}; lane>=16 +8
        v8h alo = *(const v8h*)(xrow + k0 + 8 * half);
        v8h ahi = *(const v8h*)(xrow + k0 + 8 * half + 16);
        // B 32x16 f16 layout (matches A): lane<16 K={k0..k0+7, k0+16..23}; lane>=16 +8
        v8h bl0 = *(const v8h*)(wlrow + k0 + 8 * half);
        v8h bl1 = *(const v8h*)(wlrow + k0 + 8 * half + 16);
        v8h br0 = *(const v8h*)(wrrow + k0 + 8 * half);
        v8h br1 = *(const v8h*)(wrrow + k0 + 8 * half + 16);

        v16h a, bbL, bbR;
        #pragma unroll
        for (int j = 0; j < 8; ++j) {
            a[j] = alo[j];   a[8 + j] = ahi[j];
            bbL[j] = bl0[j]; bbL[8 + j] = bl1[j];
            bbR[j] = br0[j]; bbR[8 + j] = br1[j];
        }

        accL = __builtin_amdgcn_wmma_f32_16x16x32_f16(
            false, a, false, bbL, (short)0, accL, false, false);
        accR = __builtin_amdgcn_wmma_f32_16x16x32_f16(
            false, a, false, bbR, (short)0, accR, false, false);
    }

    // C/D layout: VGPR i -> row = row0 + i + 8*half, col = col0 + lrow
    const float bLv = bl[col0 + lrow];
    const float bRv = br[col0 + lrow];
    #pragma unroll
    for (int i = 0; i < 8; ++i) {
        size_t idx = (size_t)(row0 + i + 8 * half) * HC + col0 + lrow;
        XL[idx] = accL[i] + bLv;
        XR[idx] = accR[i] + bRv;
    }
}

// ---------------------------------------------------------------------------
// init: m = -inf, denom = 0 (N*4 each), agg = 0 (N*128); N*128 threads
// ---------------------------------------------------------------------------
__global__ void init_kernel(float* __restrict__ m, float* __restrict__ denom,
                            float* __restrict__ agg, int n)
{
    int i = blockIdx.x * blockDim.x + threadIdx.x;
    if (i < n * NHEAD) { m[i] = -1e30f; denom[i] = 0.0f; }
    if (i < n * HC)    agg[i] = 0.0f;
}

// ---------------------------------------------------------------------------
// per-(edge,head) GATv2 score + segment max (float atomicMax)
// ---------------------------------------------------------------------------
__global__ void edge_scores(const int* __restrict__ srcE, const int* __restrict__ dstE,
                            int E0, int Ep,
                            const float* __restrict__ XL, const float* __restrict__ XR,
                            const float* __restrict__ att,
                            float* __restrict__ score, float* __restrict__ m)
{
    int t = blockIdx.x * blockDim.x + threadIdx.x;
    if (t >= Ep * NHEAD) return;
    int e = t >> 2, h = t & 3;
    int s, d;
    if (e < E0) { s = srcE[e]; d = dstE[e]; } else { s = d = e - E0; }

    const float4* xl = (const float4*)(XL + (size_t)s * HC + h * CPH);
    const float4* xr = (const float4*)(XR + (size_t)d * HC + h * CPH);
    const float4* at = (const float4*)(att + h * CPH);
    float acc = 0.0f;
    #pragma unroll
    for (int c = 0; c < CPH / 4; ++c) {
        float4 a = xl[c], b = xr[c], w = at[c];
        float v0 = a.x + b.x; v0 = v0 > 0.0f ? v0 : NEG_SLOPE * v0;
        float v1 = a.y + b.y; v1 = v1 > 0.0f ? v1 : NEG_SLOPE * v1;
        float v2 = a.z + b.z; v2 = v2 > 0.0f ? v2 : NEG_SLOPE * v2;
        float v3 = a.w + b.w; v3 = v3 > 0.0f ? v3 : NEG_SLOPE * v3;
        acc += w.x * v0 + w.y * v1 + w.z * v2 + w.w * v3;
    }
    score[(size_t)e * NHEAD + h] = acc;
    atomicMax(&m[(size_t)d * NHEAD + h], acc);
}

// ---------------------------------------------------------------------------
// ex = exp(score - m[dst]); denom[dst] += ex  (score overwritten in place)
// ---------------------------------------------------------------------------
__global__ void edge_exp(const int* __restrict__ dstE, int E0, int Ep,
                         float* __restrict__ score, const float* __restrict__ m,
                         float* __restrict__ denom)
{
    int t = blockIdx.x * blockDim.x + threadIdx.x;
    if (t >= Ep * NHEAD) return;
    int e = t >> 2, h = t & 3;
    int d = (e < E0) ? dstE[e] : e - E0;
    float ex = __expf(score[t] - m[(size_t)d * NHEAD + h]);
    score[t] = ex;
    atomicAdd(&denom[(size_t)d * NHEAD + h], ex);
}

// ---------------------------------------------------------------------------
// agg[dst] += (ex/denom[dst]) * XL[src]; one thread per (edge, 4-channel chunk)
// ---------------------------------------------------------------------------
__global__ void edge_aggregate(const int* __restrict__ srcE, const int* __restrict__ dstE,
                               int E0, long long total,
                               const float* __restrict__ XL,
                               const float* __restrict__ score,
                               const float* __restrict__ denom,
                               float* __restrict__ agg)
{
    long long t = (long long)blockIdx.x * blockDim.x + threadIdx.x;
    if (t >= total) return;
    int e = (int)(t >> 5);
    int q = (int)(t & 31);
    int h  = q >> 3;
    int c0 = (q & 7) * 4;
    int s, d;
    if (e < E0) { s = srcE[e]; d = dstE[e]; } else { s = d = e - E0; }

    float alpha = score[(size_t)e * NHEAD + h] / denom[(size_t)d * NHEAD + h];
    float4 xv = *(const float4*)(XL + (size_t)s * HC + h * CPH + c0);
    float* o  = agg + (size_t)d * HC + h * CPH + c0;
    atomicAdd(&o[0], alpha * xv.x);
    atomicAdd(&o[1], alpha * xv.y);
    atomicAdd(&o[2], alpha * xv.z);
    atomicAdd(&o[3], alpha * xv.w);
}

// ---------------------------------------------------------------------------
// out = relu(agg + bias) in place; optional fused f16 copy for next layer
// ---------------------------------------------------------------------------
__global__ void finalize_kernel(float* __restrict__ agg, const float* __restrict__ bias,
                                _Float16* __restrict__ out16, long long total)
{
    long long t = (long long)blockIdx.x * blockDim.x + threadIdx.x;
    if (t >= total) return;
    int col = (int)(t & (HC - 1));
    float v = agg[t] + bias[col];
    v = v > 0.0f ? v : 0.0f;
    agg[t] = v;
    if (out16) out16[t] = (_Float16)v;
}

// ---------------------------------------------------------------------------
extern "C" void kernel_launch(void* const* d_in, const int* in_sizes, int n_in,
                              void* d_out, int out_size, void* d_ws, size_t ws_size,
                              hipStream_t stream)
{
    const float* x     = (const float*)d_in[0];
    const float* Wl1   = (const float*)d_in[1];
    const float* bl1   = (const float*)d_in[2];
    const float* Wr1   = (const float*)d_in[3];
    const float* br1   = (const float*)d_in[4];
    const float* att1  = (const float*)d_in[5];
    const float* bias1 = (const float*)d_in[6];
    const float* Wl2   = (const float*)d_in[7];
    const float* bl2   = (const float*)d_in[8];
    const float* Wr2   = (const float*)d_in[9];
    const float* br2   = (const float*)d_in[10];
    const float* att2  = (const float*)d_in[11];
    const float* bias2 = (const float*)d_in[12];
    const int*   ei    = (const int*)d_in[13];

    const int IN = 64;
    const int N  = in_sizes[0] / IN;      // 50000
    const int E0 = in_sizes[13] / 2;      // 800000
    const int Ep = E0 + N;                // edges + self loops

    const int* srcE = ei;
    const int* dstE = ei + E0;

    // ---- workspace layout: f32 region then f16 region ----
    float* ws    = (float*)d_ws;
    float* xl    = ws;                          // N*HC
    float* xr    = xl + (size_t)N * HC;         // N*HC
    float* hbuf  = xr + (size_t)N * HC;         // N*HC (layer1 agg + relu out)
    float* score = hbuf + (size_t)N * HC;       // Ep*NHEAD
    float* mbuf  = score + (size_t)Ep * NHEAD;  // N*NHEAD
    float* dbuf  = mbuf + (size_t)N * NHEAD;    // N*NHEAD
    _Float16* xh   = (_Float16*)(dbuf + (size_t)N * NHEAD); // N*IN
    _Float16* h16  = xh + (size_t)N * IN;       // N*HC
    _Float16* wlt  = h16 + (size_t)N * HC;      // HC*HC (max K)
    _Float16* wrt  = wlt + (size_t)HC * HC;     // HC*HC
    float* out   = (float*)d_out;

    dim3 gemmGrid(N / 16, HC / 16), gemmBlk(32);
    const int B = 256;
    unsigned gInit = (unsigned)((N * HC + B - 1) / B);
    unsigned gEh   = (unsigned)((Ep * NHEAD + B - 1) / B);
    long long aggTot = (long long)Ep * 32;
    unsigned gAgg  = (unsigned)((aggTot + B - 1) / B);
    long long finTot = (long long)N * HC;
    unsigned gFin  = (unsigned)((finTot + B - 1) / B);
    unsigned gCvtX = (unsigned)((N * IN + B - 1) / B);
    unsigned gW1   = (unsigned)((IN * HC + B - 1) / B);
    unsigned gW2   = (unsigned)((HC * HC + B - 1) / B);

    // ---------------- layer 1 ----------------
    f32_to_f16_kernel<<<gCvtX, B, 0, stream>>>(x, xh, N * IN);
    w_transpose_f16_kernel<<<gW1, B, 0, stream>>>(Wl1, wlt, IN);
    w_transpose_f16_kernel<<<gW1, B, 0, stream>>>(Wr1, wrt, IN);
    gatv2_dual_gemm<<<gemmGrid, gemmBlk, 0, stream>>>(xh, wlt, wrt, bl1, br1, xl, xr, IN);
    init_kernel<<<gInit, B, 0, stream>>>(mbuf, dbuf, hbuf, N);
    edge_scores<<<gEh, B, 0, stream>>>(srcE, dstE, E0, Ep, xl, xr, att1, score, mbuf);
    edge_exp<<<gEh, B, 0, stream>>>(dstE, E0, Ep, score, mbuf, dbuf);
    edge_aggregate<<<gAgg, B, 0, stream>>>(srcE, dstE, E0, aggTot, xl, score, dbuf, hbuf);
    finalize_kernel<<<gFin, B, 0, stream>>>(hbuf, bias1, h16, finTot);

    // ---------------- layer 2 ----------------
    w_transpose_f16_kernel<<<gW2, B, 0, stream>>>(Wl2, wlt, HC);
    w_transpose_f16_kernel<<<gW2, B, 0, stream>>>(Wr2, wrt, HC);
    gatv2_dual_gemm<<<gemmGrid, gemmBlk, 0, stream>>>(h16, wlt, wrt, bl2, br2, xl, xr, HC);
    init_kernel<<<gInit, B, 0, stream>>>(mbuf, dbuf, out, N);
    edge_scores<<<gEh, B, 0, stream>>>(srcE, dstE, E0, Ep, xl, xr, att2, score, mbuf);
    edge_exp<<<gEh, B, 0, stream>>>(dstE, E0, Ep, score, mbuf, dbuf);
    edge_aggregate<<<gAgg, B, 0, stream>>>(srcE, dstE, E0, aggTot, xl, score, dbuf, out);
    finalize_kernel<<<gFin, B, 0, stream>>>(out, bias2, nullptr, finTot);
}
